// FusionLayer_46239617909541
// MI455X (gfx1250) — matrix operand order repaired and checked by reading, not compiled, not run on hardware
//
#include <hip/hip_runtime.h>

// ---------------------------------------------------------------------------
// CDNA5 (gfx1250, wave32) bf16 WMMA GEMM pipeline for the fusion layer:
//   k   = kmer @ Wk^T + bk            (A: f32 -> bf16 on load)
//   c   = cov  @ Wc^T + bc            (A: f32 -> bf16 on load)
//   v   = c    @ Wv^T + bv            (A: bf16 from ws)
//   att = v    @ Wo^T + bo            (A: bf16 from ws)
//   y   = relu(BN([k|att] @ Wf^T + bf))  (split-K concat, f32 out)
//
// Software-pipelined + double-buffered LDS: one barrier per 32-deep K step,
// next tile's global loads in flight behind 8 WMMAs. K must be a multiple of
// 64 (true for all five GEMMs: 4096 / 1024 / 2048 / 2048 / 4096).
// ---------------------------------------------------------------------------

typedef __attribute__((ext_vector_type(16))) __bf16 v16bf;
typedef __attribute__((ext_vector_type(8)))  __bf16 v8bf;
typedef __attribute__((ext_vector_type(4)))  __bf16 v4bf;
typedef __attribute__((ext_vector_type(8)))  float  v8f;

#define BLK        128   // block tile M and N
#define BK         32    // K step (one bf16 WMMA depth)
#define LDS_STRIDE 40    // 32 + 8 pad (bf16 elems); 80B row stride -> conflict-free
#define BUF_ELEMS  (BLK * LDS_STRIDE)

__device__ __forceinline__ v16bf cat8(v8bf lo, v8bf hi) {
  return __builtin_shufflevector(lo, hi, 0, 1, 2, 3, 4, 5, 6, 7,
                                         8, 9, 10, 11, 12, 13, 14, 15);
}

// A0/A1 + K1 implement the [k | att] concat for the final GEMM: K-column
// kk < K1 comes from A0 (row stride K1), else from A1 (row stride K-K1).
// K1 is a multiple of BK, so the select is UNIFORM per K step.
// For ordinary GEMMs pass A1 == A0 and K1 == K.
template <typename AT, bool FINAL>
__global__ __launch_bounds__(256) void gemm_wmma_kernel(
    const AT* __restrict__ A0, const AT* __restrict__ A1, int K1,
    const float* __restrict__ W,      // [N, K] row-major (so C = A @ W^T)
    const float* __restrict__ bias,   // [N]
    int M, int N, int K,
    void* __restrict__ Cout,          // bf16 [M,N] if !FINAL, f32 [M,N] if FINAL
    const float* __restrict__ gamma, const float* __restrict__ beta,
    const float* __restrict__ mean,  const float* __restrict__ var) {
  __shared__ __align__(16) __bf16 ldsA[2 * BUF_ELEMS];  // double buffered
  __shared__ __align__(16) __bf16 ldsB[2 * BUF_ELEMS];

  const int tid   = threadIdx.x;
  const int lane  = tid & 31;
  const int wave  = tid >> 5;      // 8 waves
  const int waveM = wave & 3;      // 4 waves along M
  const int waveN = wave >> 2;     // 2 waves along N
  const int mBase = blockIdx.y * BLK;
  const int nBase = blockIdx.x * BLK;
  const int wm    = waveM * 32;    // wave tile: 32 (M) x 64 (N)
  const int wn    = waveN * 64;

  // ---- fixed staging geometry: thread handles rows r0+32j, cols [c4,c4+3] ----
  const int c4   = (tid & 7) << 2;
  const int r0   = tid >> 3;
  const int lda0 = K1;
  const int lda1 = K - K1;

  const AT*    pa0[4];
  const AT*    pa1[4];
  const float* pw[4];
  int          lofs[4];
#pragma unroll
  for (int j = 0; j < 4; ++j) {
    const int r = r0 + 32 * j;
    pa0[j]  = A0 + (size_t)(mBase + r) * lda0 + c4;
    pa1[j]  = A1 + (size_t)(mBase + r) * lda1 + c4;
    pw[j]   = W + (size_t)(nBase + r) * K + c4;
    lofs[j] = r * LDS_STRIDE + c4;
  }

  // ---- fixed LDS fragment addresses (ISA Sec 7.12.2 layouts) ----
  // A 16x32: lane<16 -> row M=lane, K {0..7,16..23}; lane>=16 -> K {8..15,24..31}
  // B 32x16: lane holds column N=(lane&15); lane<16 -> K {0..7,16..23}; lane>=16 -> K {8..15,24..31}
  const __bf16* fragA = &ldsA[(wm + (lane & 15)) * LDS_STRIDE + ((lane >> 4) << 3)];
  const __bf16* fragB = &ldsB[(wn + (lane & 15)) * LDS_STRIDE + ((lane >> 4) << 3)];

  float4 aF[4];  // f32-source staging regs
  v4bf   aB[4];  // bf16-source staging regs
  float4 wF[4];  // weight staging regs (always f32 source)

  auto loadTile = [&](int kb) {
    const bool inA0 = (kb < K1);      // uniform
    const int  kk   = inA0 ? kb : kb - K1;
#pragma unroll
    for (int j = 0; j < 4; ++j) {
      const AT* p = (inA0 ? pa0[j] : pa1[j]) + kk;
      if constexpr (sizeof(AT) == 4) aF[j] = *(const float4*)p;
      else                           aB[j] = *(const v4bf*)p;
      wF[j] = *(const float4*)(pw[j] + kb);
    }
  };

  auto storeTile = [&](int buf) {   // buf is always a literal -> imm offsets
    const int bo_ = buf * BUF_ELEMS;
#pragma unroll
    for (int j = 0; j < 4; ++j) {
      v4bf a;
      if constexpr (sizeof(AT) == 4) {
        a[0] = (__bf16)aF[j].x; a[1] = (__bf16)aF[j].y;
        a[2] = (__bf16)aF[j].z; a[3] = (__bf16)aF[j].w;
      } else {
        a = aB[j];
      }
      *(v4bf*)&ldsA[bo_ + lofs[j]] = a;
      v4bf w;
      w[0] = (__bf16)wF[j].x; w[1] = (__bf16)wF[j].y;
      w[2] = (__bf16)wF[j].z; w[3] = (__bf16)wF[j].w;
      *(v4bf*)&ldsB[bo_ + lofs[j]] = w;
    }
  };

  v8f acc[2][4];
#pragma unroll
  for (int mi = 0; mi < 2; ++mi)
#pragma unroll
    for (int ni = 0; ni < 4; ++ni) acc[mi][ni] = (v8f){0, 0, 0, 0, 0, 0, 0, 0};

  auto computeTile = [&](int buf) {  // buf is always a literal
    const int bo_ = buf * BUF_ELEMS;
    v16bf af[2], bfm[4];
#pragma unroll
    for (int mi = 0; mi < 2; ++mi) {
      const __bf16* p = fragA + bo_ + mi * 16 * LDS_STRIDE;
      af[mi] = cat8(*(const v8bf*)p, *(const v8bf*)(p + 16));
    }
#pragma unroll
    for (int ni = 0; ni < 4; ++ni) {
      const __bf16* p = fragB + bo_ + ni * 16 * LDS_STRIDE;
      bfm[ni] = cat8(*(const v8bf*)p, *(const v8bf*)(p + 16));
    }
#pragma unroll
    for (int mi = 0; mi < 2; ++mi)
#pragma unroll
      for (int ni = 0; ni < 4; ++ni)
        acc[mi][ni] = __builtin_amdgcn_wmma_f32_16x16x32_bf16(
            false, af[mi], false, bfm[ni], (short)0, acc[mi][ni],
            false, false);
  };

  loadTile(0);  // prologue

  for (int kb = 0; kb < K; kb += 2 * BK) {
    // ---- stage 0: buffer 0 ----
    storeTile(0);
    __syncthreads();
    loadTile(kb + BK);                      // K % 64 == 0 -> always in range
    computeTile(0);
    // ---- stage 1: buffer 1 ----
    storeTile(1);
    __syncthreads();
    if (kb + 2 * BK < K) loadTile(kb + 2 * BK);
    computeTile(1);
  }

  // ---- epilogue: C/D layout -> lane<16: VGPR r = M row r; lane>=16: M row 8+r ----
  const int rowSel = (lane >> 4) << 3;  // 0 or 8
  const int ncol   = lane & 15;
#pragma unroll
  for (int ni = 0; ni < 4; ++ni) {
    const int n    = nBase + wn + ni * 16 + ncol;
    const float bb = bias[n];
    float g = 0.f, bt = 0.f, mu = 0.f, iv = 0.f;
    if constexpr (FINAL) {
      g  = gamma[n];
      bt = beta[n];
      mu = mean[n];
      iv = rsqrtf(var[n] + 1e-5f);
    }
#pragma unroll
    for (int mi = 0; mi < 2; ++mi) {
#pragma unroll
      for (int r = 0; r < 8; ++r) {
        const int m = mBase + wm + mi * 16 + rowSel + r;
        float val   = acc[mi][ni][r] + bb;
        if constexpr (FINAL) {
          val = (val - mu) * iv * g + bt;
          val = fmaxf(val, 0.f);
          ((float*)Cout)[(size_t)m * N + n] = val;
        } else {
          ((__bf16*)Cout)[(size_t)m * N + n] = (__bf16)val;
        }
      }
    }
  }
}

extern "C" void kernel_launch(void* const* d_in, const int* in_sizes, int n_in,
                              void* d_out, int out_size, void* d_ws,
                              size_t ws_size, hipStream_t stream) {
  const float* kmer  = (const float*)d_in[0];
  const float* cov   = (const float*)d_in[1];
  const float* Wk    = (const float*)d_in[2];
  const float* bk    = (const float*)d_in[3];
  const float* Wc    = (const float*)d_in[4];
  const float* bc    = (const float*)d_in[5];
  const float* Wv    = (const float*)d_in[6];
  const float* bv    = (const float*)d_in[7];
  const float* Wo    = (const float*)d_in[8];
  const float* bo    = (const float*)d_in[9];
  const float* Wf    = (const float*)d_in[10];
  const float* bfb   = (const float*)d_in[11];
  const float* gamma = (const float*)d_in[12];
  const float* beta  = (const float*)d_in[13];
  const float* rmean = (const float*)d_in[14];
  const float* rvar  = (const float*)d_in[15];

  const int B = 16384, KMER = 4096, COV = 1024, D = 2048;

  // bf16 workspace: k | c | v  (att overwrites c once c is dead) = 192 MB
  __bf16* k_bf   = (__bf16*)d_ws;
  __bf16* c_bf   = k_bf + (size_t)B * D;
  __bf16* v_bf   = c_bf + (size_t)B * D;
  __bf16* att_bf = c_bf;

  dim3 blk(256);
  dim3 grd(D / BLK, B / BLK);  // 16 x 128 workgroups

  // k = kmer @ Wk^T + bk
  gemm_wmma_kernel<float, false><<<grd, blk, 0, stream>>>(
      kmer, kmer, KMER, Wk, bk, B, D, KMER, (void*)k_bf,
      nullptr, nullptr, nullptr, nullptr);
  // c = cov @ Wc^T + bc
  gemm_wmma_kernel<float, false><<<grd, blk, 0, stream>>>(
      cov, cov, COV, Wc, bc, B, D, COV, (void*)c_bf,
      nullptr, nullptr, nullptr, nullptr);
  // v = c @ Wv^T + bv
  gemm_wmma_kernel<__bf16, false><<<grd, blk, 0, stream>>>(
      c_bf, c_bf, D, Wv, bv, B, D, D, (void*)v_bf,
      nullptr, nullptr, nullptr, nullptr);
  // att = v @ Wo^T + bo   (writes over dead c buffer)
  gemm_wmma_kernel<__bf16, false><<<grd, blk, 0, stream>>>(
      v_bf, v_bf, D, Wo, bo, B, D, D, (void*)att_bf,
      nullptr, nullptr, nullptr, nullptr);
  // y = relu(BN([k|att] @ Wf^T + bf))
  gemm_wmma_kernel<__bf16, true><<<grd, blk, 0, stream>>>(
      k_bf, att_bf, D, Wf, bfb, B, D, 2 * D, d_out,
      gamma, beta, rmean, rvar);
}
